// CasualAttention_39170101739775
// MI455X (gfx1250) — matrix operand order repaired and checked by reading, not compiled, not run on hardware
//
#include <hip/hip_runtime.h>

// ---------------- problem constants ----------------
static constexpr int Bb = 4;
static constexpr int Tt = 2048;
static constexpr int Cc = 1024;
static constexpr int Hh = 16;
static constexpr int Dd = 64;

// ---------------- WMMA types ----------------
typedef __attribute__((ext_vector_type(16))) __bf16 bf16x16;
typedef __attribute__((ext_vector_type(8)))  float  f32x8;

union Frag {
    bf16x16 v;
    uint4   q[2];
};

__device__ __forceinline__ unsigned short f32_to_bf16(float f) {
    union { float f; unsigned int u; } v;
    v.f = f;
    unsigned int u = v.u;
    unsigned int rounding = 0x7FFFu + ((u >> 16) & 1u);
    return (unsigned short)((u + rounding) >> 16);
}

// A fragment (16x32 bf16, row-major source [m][k], ld in elements).
// Contiguous k order: lane m = lane&15; lanes0-15 hold K0-15, lanes16-31 hold K16-31.
__device__ __forceinline__ bf16x16 load_frag_a(const unsigned short* base, int ld,
                                               int mBase, int kOff, int lane) {
    const int m  = mBase + (lane & 15);
    const int kh = (lane >> 4) << 4;  // 0 or 16
    const unsigned short* p = base + (size_t)m * ld + kOff + kh;
    Frag f;
    f.q[0] = *(const uint4*)(p);
    f.q[1] = *(const uint4*)(p + 8);
    return f.v;
}

// B fragment (32x16 bf16) from transposed storage [n][k] (ld in elements).
// Contiguous k order: lane n = lane&15; lanes0-15 hold K0-15, lanes16-31 hold K16-31.
__device__ __forceinline__ bf16x16 load_frag_b(const unsigned short* base, int ld,
                                               int nBase, int kOff, int lane) {
    const int n  = nBase + (lane & 15);
    const int kh = (lane >> 4) << 4;  // 0 or 16
    const unsigned short* p = base + (size_t)n * ld + kOff + kh;
    Frag f;
    f.q[0] = *(const uint4*)(p);
    f.q[1] = *(const uint4*)(p + 8);
    return f.v;
}

__device__ __forceinline__ f32x8 wmma_bf16(bf16x16 a, bf16x16 b, f32x8 c) {
    return __builtin_amdgcn_wmma_f32_16x16x32_bf16(false, a, false, b,
                                                   (short)0, c, false, false);
}

// ---------------- CDNA5 async global->LDS copy (ASYNCcnt path) ----------------
// Each active lane copies 16 bytes from its global address to its LDS offset.
// LDS byte offset = low 32 bits of the generic flat address (ISA 10.2: the
// LDS aperture occupies addr[63:32]).
__device__ __forceinline__ unsigned lds_off(const void* p) {
    return (unsigned)(uintptr_t)p;
}
__device__ __forceinline__ void async_copy_b128(unsigned ldsaddr, const void* gaddr) {
    asm volatile("global_load_async_to_lds_b128 %0, %1, off"
                 :: "v"(ldsaddr), "v"(gaddr)
                 : "memory");
}
__device__ __forceinline__ void wait_async_le1() {
    asm volatile("s_wait_asynccnt 0x1" ::: "memory");
}
__device__ __forceinline__ void wait_async_le0() {
    asm volatile("s_wait_asynccnt 0x0" ::: "memory");
}

// ---------------- conversion kernels ----------------
__global__ void cvt_f32_bf16(const float* __restrict__ src,
                             unsigned short* __restrict__ dst, int n) {
    int i = blockIdx.x * 256 + threadIdx.x;
    if (i < n) dst[i] = f32_to_bf16(src[i]);
}

// src [K][N] f32 row-major -> dst [N][K] bf16 (transposed)
__global__ void transpose_cvt(const float* __restrict__ src,
                              unsigned short* __restrict__ dst, int K, int N) {
    int i = blockIdx.x * 256 + threadIdx.x;
    if (i < K * N) {
        int k = i / N;
        int n = i - k * N;
        dst[(size_t)n * K + k] = f32_to_bf16(src[i]);
    }
}

// ---------------- QKV projection GEMM + scatter ----------------
// A: xb [M=B*T][K=C] bf16, BT: wT [3C][C] bf16, bias [3C] f32.
// Block: 256 threads = 8 waves; block tile 256(M) x 64(N); wave tile 32x64.
// B tile (shared by all 8 waves) is staged via async global->LDS, 2-deep pipeline.
__global__ __launch_bounds__(256) void gemm_qkv(const unsigned short* __restrict__ A,
                                                const unsigned short* __restrict__ BT,
                                                const float* __restrict__ bias,
                                                unsigned short* __restrict__ qs,
                                                unsigned short* __restrict__ ks,
                                                unsigned short* __restrict__ vs) {
    __shared__ __align__(16) unsigned short Bs[2][64 * 32];

    const int lane  = threadIdx.x & 31;
    const int wave  = threadIdx.x >> 5;
    const int mBase = blockIdx.y * 256 + wave * 32;
    const int nBase = blockIdx.x * 64;

    // async staging: thread -> one 16B chunk of the 64x32 B tile
    const int sRow   = threadIdx.x >> 2;        // 0..63 (n within tile)
    const int sChunk = (threadIdx.x & 3) * 8;   // element offset in k
    const unsigned short* gB = BT + (size_t)(nBase + sRow) * Cc + sChunk;
    const unsigned ldsB0 = lds_off(&Bs[0][sRow * 32 + sChunk]);
    const unsigned ldsB1 = lds_off(&Bs[1][sRow * 32 + sChunk]);

    f32x8 acc[8] = {f32x8{}, f32x8{}, f32x8{}, f32x8{},
                    f32x8{}, f32x8{}, f32x8{}, f32x8{}};

    // prologue: stage k0 = 0 into buffer 0
    async_copy_b128(ldsB0, gB);

    for (int k0 = 0, it = 0; k0 < Cc; k0 += 32, ++it) {
        const int  cur  = it & 1;
        const bool more = (k0 + 32) < Cc;

        __syncthreads();   // all waves done reading buf[1-cur] from prev iter
        if (more) async_copy_b128(cur ? ldsB0 : ldsB1, gB + k0 + 32);
        if (more) wait_async_le1(); else wait_async_le0();  // buf[cur] landed
        __syncthreads();   // visible to all waves

        __builtin_prefetch(A + (size_t)mBase * Cc + k0 + 64, 0, 0);
        bf16x16 a0 = load_frag_a(A, Cc, mBase,      k0, lane);
        bf16x16 a1 = load_frag_a(A, Cc, mBase + 16, k0, lane);
        const unsigned short* bs = Bs[cur];
#pragma unroll
        for (int nt = 0; nt < 4; ++nt) {
            bf16x16 b = load_frag_b(bs, 32, nt * 16, 0, lane);
            acc[nt]     = wmma_bf16(a0, b, acc[nt]);
            acc[4 + nt] = wmma_bf16(a1, b, acc[4 + nt]);
        }
    }

    const int colL   = lane & 15;
    const int rowOff = (lane >> 4) << 3;
#pragma unroll
    for (int rg = 0; rg < 2; ++rg) {
#pragma unroll
        for (int nt = 0; nt < 4; ++nt) {
            int n   = nBase + nt * 16 + colL;
            int sel = n >> 10;            // 0=q 1=k 2=v
            int h   = (n >> 6) & 15;
            int d   = n & 63;
            unsigned short* dst = (sel == 0) ? qs : (sel == 1) ? ks : vs;
            float bv = bias[n];
#pragma unroll
            for (int r = 0; r < 8; ++r) {
                int m = mBase + rg * 16 + r + rowOff;
                int b = m >> 11;          // /T
                int t = m & (Tt - 1);
                dst[(((size_t)(b * Hh + h) * Tt) + t) * Dd + d] =
                    f32_to_bf16(acc[rg * 4 + nt][r] + bv);
            }
        }
    }
}

// ---------------- flash attention ----------------
// grid: (T/128, B*H); block 256 = 8 waves; wave -> 16 query rows.
__global__ __launch_bounds__(256) void flash_attn(const unsigned short* __restrict__ qs,
                                                  const unsigned short* __restrict__ ks,
                                                  const unsigned short* __restrict__ vs,
                                                  unsigned short* __restrict__ ob) {
    __shared__ __align__(16) unsigned short Vt[Dd * 32];        // [d][key] transposed V block
    __shared__ __align__(16) unsigned short Pl[8 * 16 * 32];    // per-wave P tiles

    const int lane  = threadIdx.x & 31;
    const int wave  = threadIdx.x >> 5;
    const int bh    = blockIdx.y;
    const int b     = bh >> 4;
    const int h     = bh & 15;
    const int qBase = blockIdx.x * 128 + wave * 16;

    const unsigned short* Q = qs + (size_t)bh * Tt * Dd;
    const unsigned short* K = ks + (size_t)bh * Tt * Dd;
    const unsigned short* V = vs + (size_t)bh * Tt * Dd;

    // Q fragments cached in registers for the whole key loop
    bf16x16 qf0 = load_frag_a(Q, Dd, qBase, 0, lane);
    bf16x16 qf1 = load_frag_a(Q, Dd, qBase, 32, lane);

    f32x8 acc[4] = {f32x8{}, f32x8{}, f32x8{}, f32x8{}};
    float mrun[8], lrun[8];
#pragma unroll
    for (int r = 0; r < 8; ++r) { mrun[r] = -1.0e30f; lrun[r] = 0.0f; }

    const int colL   = lane & 15;
    const int rowOff = (lane >> 4) << 3;
    const int nkb    = blockIdx.x * 4 + 4;   // key blocks of 32 covering causal range

    for (int kb = 0; kb < nkb; ++kb) {
        const int kBase = kb * 32;

        // cooperatively stage V^T (32 keys x 64 d) into LDS
        __syncthreads();
        for (int i = threadIdx.x; i < 32 * Dd; i += 256) {
            int key = i >> 6;
            int d   = i & 63;
            Vt[d * 32 + key] = V[(size_t)(kBase + key) * Dd + d];
        }
        __syncthreads();

        if (kBase <= qBase + 15) {   // wave-uniform causal skip
            // ---- scores S = Q K^T : two 16x16 tiles over 32 keys ----
            f32x8 s0 = f32x8{}, s1 = f32x8{};
            {
                bf16x16 k00 = load_frag_b(K, Dd, kBase,      0,  lane);
                bf16x16 k01 = load_frag_b(K, Dd, kBase,      32, lane);
                s0 = wmma_bf16(qf0, k00, s0);
                s0 = wmma_bf16(qf1, k01, s0);
                bf16x16 k10 = load_frag_b(K, Dd, kBase + 16, 0,  lane);
                bf16x16 k11 = load_frag_b(K, Dd, kBase + 16, 32, lane);
                s1 = wmma_bf16(qf0, k10, s1);
                s1 = wmma_bf16(qf1, k11, s1);
            }

            // ---- causal mask + online softmax ----
#pragma unroll
            for (int r = 0; r < 8; ++r) {
                int row = qBase + r + rowOff;
                if (kBase + colL      > row) s0[r] = -1.0e30f;
                if (kBase + 16 + colL > row) s1[r] = -1.0e30f;

                float mx = fmaxf(s0[r], s1[r]);
                mx = fmaxf(mx, __shfl_xor(mx, 1, 32));
                mx = fmaxf(mx, __shfl_xor(mx, 2, 32));
                mx = fmaxf(mx, __shfl_xor(mx, 4, 32));
                mx = fmaxf(mx, __shfl_xor(mx, 8, 32));

                float mnew = fmaxf(mrun[r], mx);
                float sc   = __expf(mrun[r] - mnew);
                float p0   = __expf(s0[r] - mnew);
                float p1   = __expf(s1[r] - mnew);
                s0[r] = p0; s1[r] = p1;

                float rs = p0 + p1;
                rs += __shfl_xor(rs, 1, 32);
                rs += __shfl_xor(rs, 2, 32);
                rs += __shfl_xor(rs, 4, 32);
                rs += __shfl_xor(rs, 8, 32);

                lrun[r] = lrun[r] * sc + rs;
                mrun[r] = mnew;
                acc[0][r] *= sc; acc[1][r] *= sc;
                acc[2][r] *= sc; acc[3][r] *= sc;
            }

            // ---- reshape P (C-layout f32) -> A-fragment (bf16) via wave-private LDS ----
            unsigned short* Pw = Pl + wave * 512;
#pragma unroll
            for (int r = 0; r < 8; ++r) {
                int row = r + rowOff;
                Pw[row * 32 + colL]      = f32_to_bf16(s0[r]);
                Pw[row * 32 + 16 + colL] = f32_to_bf16(s1[r]);
            }
            // same-wave LDS ops are in-order; loads below see the stores above
            bf16x16 pf = load_frag_a(Pw, 32, 0, 0, lane);

            // ---- O += P * V ----
#pragma unroll
            for (int nt = 0; nt < 4; ++nt) {
                bf16x16 vb = load_frag_b(Vt, 32, nt * 16, 0, lane);
                acc[nt] = wmma_bf16(pf, vb, acc[nt]);
            }
        }
    }

    // epilogue: normalize and store bf16 into [B,T,C] layout
#pragma unroll
    for (int nt = 0; nt < 4; ++nt) {
#pragma unroll
        for (int r = 0; r < 8; ++r) {
            int t = qBase + r + rowOff;
            float o = acc[nt][r] / lrun[r];
            ob[((size_t)b * Tt + t) * Cc + h * Dd + nt * 16 + colL] = f32_to_bf16(o);
        }
    }
}

// ---------------- output projection GEMM ----------------
// A: attn bf16 [M=B*T][C], BT: projT [C][C] bf16, out f32 [M][C] + bias.
// Same async-staged structure as gemm_qkv.
__global__ __launch_bounds__(256) void gemm_proj(const unsigned short* __restrict__ A,
                                                 const unsigned short* __restrict__ BT,
                                                 const float* __restrict__ bias,
                                                 float* __restrict__ out) {
    __shared__ __align__(16) unsigned short Bs[2][64 * 32];

    const int lane  = threadIdx.x & 31;
    const int wave  = threadIdx.x >> 5;
    const int mBase = blockIdx.y * 256 + wave * 32;
    const int nBase = blockIdx.x * 64;

    const int sRow   = threadIdx.x >> 2;
    const int sChunk = (threadIdx.x & 3) * 8;
    const unsigned short* gB = BT + (size_t)(nBase + sRow) * Cc + sChunk;
    const unsigned ldsB0 = lds_off(&Bs[0][sRow * 32 + sChunk]);
    const unsigned ldsB1 = lds_off(&Bs[1][sRow * 32 + sChunk]);

    f32x8 acc[8] = {f32x8{}, f32x8{}, f32x8{}, f32x8{},
                    f32x8{}, f32x8{}, f32x8{}, f32x8{}};

    async_copy_b128(ldsB0, gB);

    for (int k0 = 0, it = 0; k0 < Cc; k0 += 32, ++it) {
        const int  cur  = it & 1;
        const bool more = (k0 + 32) < Cc;

        __syncthreads();
        if (more) async_copy_b128(cur ? ldsB0 : ldsB1, gB + k0 + 32);
        if (more) wait_async_le1(); else wait_async_le0();
        __syncthreads();

        __builtin_prefetch(A + (size_t)mBase * Cc + k0 + 64, 0, 0);
        bf16x16 a0 = load_frag_a(A, Cc, mBase,      k0, lane);
        bf16x16 a1 = load_frag_a(A, Cc, mBase + 16, k0, lane);
        const unsigned short* bs = Bs[cur];
#pragma unroll
        for (int nt = 0; nt < 4; ++nt) {
            bf16x16 b = load_frag_b(bs, 32, nt * 16, 0, lane);
            acc[nt]     = wmma_bf16(a0, b, acc[nt]);
            acc[4 + nt] = wmma_bf16(a1, b, acc[4 + nt]);
        }
    }

    const int colL   = lane & 15;
    const int rowOff = (lane >> 4) << 3;
#pragma unroll
    for (int rg = 0; rg < 2; ++rg) {
#pragma unroll
        for (int nt = 0; nt < 4; ++nt) {
            int n = nBase + nt * 16 + colL;
            float bv = bias[n];
#pragma unroll
            for (int r = 0; r < 8; ++r) {
                int m = mBase + rg * 16 + r + rowOff;
                out[(size_t)m * Cc + n] = acc[rg * 4 + nt][r] + bv;
            }
        }
    }
}

// ---------------- launcher ----------------
extern "C" void kernel_launch(void* const* d_in, const int* in_sizes, int n_in,
                              void* d_out, int out_size, void* d_ws, size_t ws_size,
                              hipStream_t stream) {
    const float* x      = (const float*)d_in[0];   // [B,T,C]
    const float* attn_w = (const float*)d_in[1];   // [C,3C]
    const float* attn_b = (const float*)d_in[2];   // [3C]
    const float* proj_w = (const float*)d_in[3];   // [C,C]
    const float* proj_b = (const float*)d_in[4];   // [C]
    float* out = (float*)d_out;                    // [B,T,C] f32

    const size_t M = (size_t)Bb * Tt;              // 8192

    // workspace carve-out (bf16 buffers), 16B aligned
    char* ws = (char*)d_ws;
    size_t off = 0;
    unsigned short* xb     = (unsigned short*)(ws + off); off += M * Cc * 2;
    unsigned short* wqkvT  = (unsigned short*)(ws + off); off += (size_t)3 * Cc * Cc * 2;
    unsigned short* wprojT = (unsigned short*)(ws + off); off += (size_t)Cc * Cc * 2;
    unsigned short* q_s    = (unsigned short*)(ws + off); off += M * Cc * 2;
    unsigned short* k_s    = (unsigned short*)(ws + off); off += M * Cc * 2;
    unsigned short* v_s    = (unsigned short*)(ws + off); off += M * Cc * 2;
    unsigned short* att_b  = (unsigned short*)(ws + off); off += M * Cc * 2;

    // 1) conversions / weight transposes
    {
        int n = (int)(M * Cc);
        cvt_f32_bf16<<<(n + 255) / 256, 256, 0, stream>>>(x, xb, n);
        int nw = Cc * 3 * Cc;
        transpose_cvt<<<(nw + 255) / 256, 256, 0, stream>>>(attn_w, wqkvT, Cc, 3 * Cc);
        int np = Cc * Cc;
        transpose_cvt<<<(np + 255) / 256, 256, 0, stream>>>(proj_w, wprojT, Cc, Cc);
    }

    // 2) QKV projection (WMMA + async LDS staging) with scatter to [B,H,T,D]
    gemm_qkv<<<dim3(3 * Cc / 64, (int)(M / 256)), 256, 0, stream>>>(
        xb, wqkvT, attn_b, q_s, k_s, v_s);

    // 3) causal flash attention (WMMA + LDS)
    flash_attn<<<dim3(Tt / 128, Bb * Hh), 256, 0, stream>>>(q_s, k_s, v_s, att_b);

    // 4) output projection (WMMA + async LDS staging) to fp32 + bias
    gemm_proj<<<dim3(Cc / 64, (int)(M / 256)), 256, 0, stream>>>(
        att_b, wprojT, proj_b, out);
}
